// LiquidCell_46359876993583
// MI455X (gfx1250) — matrix operand order repaired and checked by reading, not compiled, not run on hardware
//
#include <hip/hip_runtime.h>
#include <hip/hip_bf16.h>

#define TT 512
#define BB 256
#define IN 128
#define HH 512
#define ZK 640   // IN + HH
#define TAU_MIN 0.2f
#define TAU_MAX 2.0f
#define DT_C    0.1f

typedef __bf16 v16bf __attribute__((ext_vector_type(16)));
typedef float  v8f   __attribute__((ext_vector_type(8)));

union FragU { int4 q[2]; unsigned short s[16]; v16bf v; };

__device__ __forceinline__ unsigned short f2bf(float f) {
    unsigned int u = __float_as_uint(f);
    unsigned int r = (u + 0x7FFFu + ((u >> 16) & 1u)) >> 16;  // round-to-nearest-even
    return (unsigned short)r;
}

// Zero h0 (fp32 + bf16 mirror) and convert+transpose weights: WT[n*ZK + k] = W[k*HH + n]
__global__ void ltc_init(const float* __restrict__ Wc, const float* __restrict__ Wt,
                         unsigned short* __restrict__ WcT, unsigned short* __restrict__ WtT,
                         float* __restrict__ h0, unsigned short* __restrict__ hbf0) {
    int idx = blockIdx.x * blockDim.x + threadIdx.x;
    if (idx < ZK * HH) {
        int k = idx / HH;
        int n = idx - k * HH;
        WcT[n * ZK + k] = f2bf(Wc[idx]);
        WtT[n * ZK + k] = f2bf(Wt[idx]);
    }
    if (idx < BB * HH) { h0[idx] = 0.0f; hbf0[idx] = 0; }
}

// One-time conversion of the whole input sequence to bf16 (off the serial path).
__global__ void ltc_pack_x(const float* __restrict__ x, unsigned short* __restrict__ Xbf,
                           int n) {
    int idx = blockIdx.x * blockDim.x + threadIdx.x;
    if (idx < n) Xbf[idx] = f2bf(x[idx]);
}

// Fused recurrence step. Block = 256 threads = 8 waves; grid (8,8):
// 32 batch rows x 64 hidden cols per block; each wave owns one 16x16 tile of
// BOTH Cc and Ct, then fuses tanh/sigmoid/ODE update, writes h_next (fp32 +
// bf16 for next step's GEMM), tau, and per-16-col y partial dot products.
__global__ void __launch_bounds__(256)
ltc_step(const unsigned short* __restrict__ xbf,       // [B][IN] bf16 (x_t)
         const unsigned short* __restrict__ hbf,       // [B][H] bf16 (h_t)
         const unsigned short* __restrict__ WcT,       // [H][ZK] bf16 (Wc^T)
         const unsigned short* __restrict__ WtT,       // [H][ZK] bf16 (Wt^T)
         const float* __restrict__ bc, const float* __restrict__ bt,
         const float* __restrict__ h_cur,              // [B][H] fp32 (h_t)
         const float* __restrict__ Wo,                 // [H]
         float* __restrict__ h_next,                   // [B][H] fp32
         unsigned short* __restrict__ hbf_next,        // [B][H] bf16
         float* __restrict__ tau_out,                  // d_out tau slice [B][H]
         float* __restrict__ ypart)                    // [32][B] partial dots
{
    const int lane = threadIdx.x & 31;
    const int wave = threadIdx.x >> 5;
    const int r = wave & 1;          // row subtile 0..1
    const int c = wave >> 1;         // col subtile 0..3
    const int rowBase = blockIdx.x * 32 + r * 16;
    const int colBase = blockIdx.y * 64 + c * 16;

    const int l16 = lane & 15;
    const int hiHalf = (lane >= 16);
    const int arow = rowBase + l16;
    const int bcol = colBase + l16;
    const int krelA = hiHalf ? 16 : 0;   // A: K 0-15 (lo lanes) vs 16-31 (hi)
    const int krelB = hiHalf ? 16 : 0;   // B: K 0-15 (lo lanes) vs 16-31 (hi)

    const unsigned short* xrow  = xbf + (size_t)arow * IN;
    const unsigned short* hrow  = hbf + (size_t)arow * HH;
    const unsigned short* wcCol = WcT + (size_t)bcol * ZK;
    const unsigned short* wtCol = WtT + (size_t)bcol * ZK;

    v8f accC = {0.f,0.f,0.f,0.f,0.f,0.f,0.f,0.f};
    v8f accT = {0.f,0.f,0.f,0.f,0.f,0.f,0.f,0.f};

    // ---- K region 0..127: A from pre-converted x_t (pure 128-bit loads) ----
#pragma unroll
    for (int kk = 0; kk < IN; kk += 32) {
        FragU a, bC, bT;
        a.q[0]  = *reinterpret_cast<const int4*>(xrow + kk + krelA);
        a.q[1]  = *reinterpret_cast<const int4*>(xrow + kk + krelA + 8);
        bC.q[0] = *reinterpret_cast<const int4*>(wcCol + kk + krelB);
        bC.q[1] = *reinterpret_cast<const int4*>(wcCol + kk + krelB + 8);
        bT.q[0] = *reinterpret_cast<const int4*>(wtCol + kk + krelB);
        bT.q[1] = *reinterpret_cast<const int4*>(wtCol + kk + krelB + 8);
        accC = __builtin_amdgcn_wmma_f32_16x16x32_bf16(false, a.v, false, bC.v,
                                                       (short)0, accC, false, false);
        accT = __builtin_amdgcn_wmma_f32_16x16x32_bf16(false, a.v, false, bT.v,
                                                       (short)0, accT, false, false);
    }

    // ---- K region 128..639: A from hbf (pure 128-bit loads, fully unrolled) ----
#pragma unroll
    for (int kk = 0; kk < HH; kk += 32) {
        FragU a, bC, bT;
        a.q[0]  = *reinterpret_cast<const int4*>(hrow + kk + krelA);
        a.q[1]  = *reinterpret_cast<const int4*>(hrow + kk + krelA + 8);
        bC.q[0] = *reinterpret_cast<const int4*>(wcCol + IN + kk + krelB);
        bC.q[1] = *reinterpret_cast<const int4*>(wcCol + IN + kk + krelB + 8);
        bT.q[0] = *reinterpret_cast<const int4*>(wtCol + IN + kk + krelB);
        bT.q[1] = *reinterpret_cast<const int4*>(wtCol + IN + kk + krelB + 8);
        accC = __builtin_amdgcn_wmma_f32_16x16x32_bf16(false, a.v, false, bC.v,
                                                       (short)0, accC, false, false);
        accT = __builtin_amdgcn_wmma_f32_16x16x32_bf16(false, a.v, false, bT.v,
                                                       (short)0, accT, false, false);
    }

    // ---- Fused epilogue ----
    // C/D layout: vgpr i, lanes 0-15 -> M=i, lanes 16-31 -> M=i+8; N = lane%16
    const int n = bcol;
    const float biasC = bc[n];
    const float biasT = bt[n];
    const float won   = Wo[n];
    const int rlo = rowBase + (hiHalf ? 8 : 0);
    float yacc[8];
#pragma unroll
    for (int i = 0; i < 8; ++i) {
        const int row = rlo + i;
        const float cand = tanhf(accC[i] + biasC);
        const float sg   = 1.0f / (1.0f + expf(-(accT[i] + biasT)));
        const float tau  = TAU_MIN + sg * (TAU_MAX - TAU_MIN);
        const float h    = h_cur[(size_t)row * HH + n];
        const float hn   = h + DT_C * (cand - h) / tau;
        h_next  [(size_t)row * HH + n] = hn;
        hbf_next[(size_t)row * HH + n] = f2bf(hn);
        tau_out [(size_t)row * HH + n] = tau;
        yacc[i] = hn * won;
    }
    // reduce across the 16 column lanes (stays within each 16-lane half)
#pragma unroll
    for (int mask = 1; mask < 16; mask <<= 1) {
#pragma unroll
        for (int i = 0; i < 8; ++i) yacc[i] += __shfl_xor(yacc[i], mask, 32);
    }
    if (l16 == 0) {
        const int chunk = blockIdx.y * 4 + c;   // 0..31 (16-col chunk index)
#pragma unroll
        for (int i = 0; i < 8; ++i) ypart[chunk * BB + rlo + i] = yacc[i];
    }
}

// One pass over all (t,b): y[t,b] = sum_{j<32} ypart[t][j][b] + bo
__global__ void ltc_y_final(const float* __restrict__ ypart,   // [T][32][B]
                            const float* __restrict__ bo,
                            float* __restrict__ y) {           // [T*B]
    int idx = blockIdx.x * blockDim.x + threadIdx.x;
    if (idx >= TT * BB) return;
    int t = idx / BB;
    int b = idx - t * BB;
    const float* p = ypart + (size_t)t * 32 * BB + b;
    float s = 0.0f;
#pragma unroll
    for (int j = 0; j < 32; ++j) s += p[j * BB];
    y[idx] = s + bo[0];
}

extern "C" void kernel_launch(void* const* d_in, const int* in_sizes, int n_in,
                              void* d_out, int out_size, void* d_ws, size_t ws_size,
                              hipStream_t stream) {
    const float* x   = (const float*)d_in[0];   // [T,B,IN]
    const float* Wc  = (const float*)d_in[1];   // [ZK,H]
    const float* bc  = (const float*)d_in[2];   // [H]
    const float* Wt  = (const float*)d_in[3];   // [ZK,H]
    const float* bt  = (const float*)d_in[4];   // [H]
    const float* Wo  = (const float*)d_in[5];   // [H]
    const float* bo  = (const float*)d_in[6];   // [1]
    float* out = (float*)d_out;                 // y_seq [T*B] then tau_seq [T*B*H]

    char* ws = (char*)d_ws;
    const size_t BH  = (size_t)BB * HH;
    const size_t NX  = (size_t)TT * BB * IN;
    float*          hbuf  = (float*)ws;                               // 2 * B*H fp32
    unsigned short* hbf   = (unsigned short*)(ws + 2 * BH * 4);       // 2 * B*H bf16
    unsigned short* WcT   = hbf + 2 * BH;                             // H*ZK bf16
    unsigned short* WtT   = WcT + (size_t)ZK * HH;
    unsigned short* Xbf   = WtT + (size_t)ZK * HH;                    // T*B*IN bf16
    float*          ypart = (float*)(Xbf + NX);                       // T*32*B fp32

    ltc_init<<<(ZK * HH + 255) / 256, 256, 0, stream>>>(Wc, Wt, WcT, WtT, hbuf, hbf);
    ltc_pack_x<<<(int)((NX + 255) / 256), 256, 0, stream>>>(x, Xbf, (int)NX);

    float* tau_base = out + (size_t)TT * BB;
    for (int t = 0; t < TT; ++t) {
        const float*          h_cur  = hbuf + (size_t)(t & 1) * BH;
        float*                h_nxt  = hbuf + (size_t)((t + 1) & 1) * BH;
        const unsigned short* hb_cur = hbf + (size_t)(t & 1) * BH;
        unsigned short*       hb_nxt = hbf + (size_t)((t + 1) & 1) * BH;
        ltc_step<<<dim3(8, 8), 256, 0, stream>>>(
            Xbf + (size_t)t * BB * IN, hb_cur, WcT, WtT, bc, bt, h_cur, Wo,
            h_nxt, hb_nxt, tau_base + (size_t)t * BB * HH,
            ypart + (size_t)t * 32 * BB);
    }
    ltc_y_final<<<(TT * BB + 255) / 256, 256, 0, stream>>>(ypart, bo, out);
}
